// NodeMultiHeadAttention_51539608256
// MI455X (gfx1250) — matrix operand order repaired and checked by reading, not compiled, not run on hardware
//
#include <hip/hip_runtime.h>

// GAT-style multi-head attention for MI455X (gfx1250), wave32 + WMMA bf16.
//
// Pipeline (all on `stream`):
//  k0: convert 7 weight matrices (128x128 fp32) -> bf16 in ws
//  k1: node projections nodeQ/K/V = nodes @ W.T + b    (WMMA bf16, fp32 acc)
//  k2: per-16-edge tile: edge Q/K/V GEMMs + gather node proj; wave w == head w
//      (DH==16 == WMMA tile width) -> score[E,8] via in-wave shuffles,
//      V[E,128] materialized with NT stores
//  k3: segment max (ordered-uint atomicMax)
//  k4: exp + segment sum (atomicAdd)
//  k5: weighted scatter-add of V into agg[N,128] (b128 NT loads, 4 atomics)
//  k6: out = agg @ Wo.T + bo  (WMMA bf16)

typedef __attribute__((ext_vector_type(16))) __bf16 v16bf;
typedef __attribute__((ext_vector_type(4)))  __bf16 v4bf;
typedef __attribute__((ext_vector_type(8)))  float  v8f;
typedef __attribute__((ext_vector_type(4)))  float  v4f;

#define Dn 128
#define Hh 8

// ---------- float <-> order-preserving uint key (for atomicMax on floats) ----
__device__ __forceinline__ unsigned fkey(float f) {
  unsigned u = __float_as_uint(f);
  return (u & 0x80000000u) ? ~u : (u | 0x80000000u);
}
__device__ __forceinline__ float funkey(unsigned k) {
  unsigned u = (k & 0x80000000u) ? (k & 0x7fffffffu) : ~k;
  return __uint_as_float(u);
}

// ---------- tile staging: 16x128 fp32 rows -> bf16 in LDS (b128 loads) -------
__device__ __forceinline__ void stage_tile_bf16(const float* __restrict__ src,
                                                int row0, int nrows, int tid,
                                                __bf16* ldsA) {
#pragma unroll
  for (int i = 0; i < 2; ++i) {
    int idx4 = tid + i * 256;              // 512 groups of 4 elements
    int r = idx4 >> 5;                     // row 0..15
    int c4 = (idx4 & 31) << 2;             // col 0,4,...,124
    int gr = row0 + r; if (gr >= nrows) gr = nrows - 1;
    v4f f = __builtin_nontemporal_load(
        (const v4f*)(src + (size_t)gr * Dn + c4));
    v4bf h;
    h[0] = (__bf16)f[0]; h[1] = (__bf16)f[1];
    h[2] = (__bf16)f[2]; h[3] = (__bf16)f[3];
    *(v4bf*)(ldsA + idx4 * 4) = h;
  }
}

// ---------- WMMA fragment helpers (16x16x32 bf16; contiguous k order used by this code) --------------
// A (16x32 bf16): lanes 0-15 row M=lane, lanes 16-31 row M=lane-16.
// half i of lane (hi=lane>>4):  K = kt*32 + hi*16 + i
__device__ __forceinline__ void build_a_frags(const __bf16* ldsA, int lane, v16bf a[4]) {
  const int r = lane & 15, hi = lane >> 4;
#pragma unroll
  for (int kt = 0; kt < 4; ++kt) {
#pragma unroll
    for (int i = 0; i < 16; ++i) {
      int k = kt * 32 + hi * 16 + i;
      a[kt][i] = ldsA[r * Dn + k];
    }
  }
}

// B (32x16 bf16): lane holds column N=lane&15; half i -> K = kt*32 + hi*16 + i.
// Weight W is [128 out][128 in] row-major bf16; B[k][n] = W[n][k] -> 16
// contiguous halves per fragment (one b256-equivalent global load pair).
// C/D (16x16 f32): c[r] = element(row = r + hi*8, col = lane&15).
__device__ __forceinline__ v8f gemm_tile(const v16bf a[4], const __bf16* Wb,
                                         int ncol, int hi, float binit) {
  v8f c;
#pragma unroll
  for (int r = 0; r < 8; ++r) c[r] = binit;
#pragma unroll
  for (int kt = 0; kt < 4; ++kt) {
    v16bf b = *(const v16bf*)(Wb + (size_t)ncol * Dn + kt * 32 + hi * 16);
    c = __builtin_amdgcn_wmma_f32_16x16x32_bf16(false, a[kt], false, b,
                                                (short)0, c, false, false);
  }
  return c;
}

// ---------- k0: weights fp32 -> bf16 ----------------------------------------
__global__ __launch_bounds__(256) void k_cvt_weights(
    const float* Wnq, const float* Wnk, const float* Wnv,
    const float* Weq, const float* Wek, const float* Wev,
    const float* Wo, __bf16* out) {
  int idx = blockIdx.x * 256 + threadIdx.x;     // 0 .. 7*16384-1
  int m = idx >> 14, off = idx & 16383;
  const float* src;
  switch (m) {
    case 0: src = Wnq; break; case 1: src = Wnk; break;
    case 2: src = Wnv; break; case 3: src = Weq; break;
    case 4: src = Wek; break; case 5: src = Wev; break;
    default: src = Wo;
  }
  out[idx] = (__bf16)src[off];
}

// ---------- k1: node projections (3 fused GEMMs sharing the A tile) ----------
__global__ __launch_bounds__(256) void k_node_proj(
    const float* __restrict__ nodes, const __bf16* __restrict__ wb16,
    const float* __restrict__ bnq, const float* __restrict__ bnk,
    const float* __restrict__ bnv,
    float* __restrict__ nq, float* __restrict__ nk, float* __restrict__ nv,
    int nrows) {
  __shared__ __attribute__((aligned(16))) __bf16 ldsA[16 * Dn];
  const int tid = threadIdx.x;
  const int row0 = blockIdx.x * 16;
  stage_tile_bf16(nodes, row0, nrows, tid, ldsA);
  __syncthreads();

  const int wave = tid >> 5, lane = tid & 31;
  const int col = lane & 15, hi = lane >> 4;
  v16bf a[4];
  build_a_frags(ldsA, lane, a);
  const int ncol = wave * 16 + col;

  v8f acc[3];
  acc[0] = gemm_tile(a, wb16,                     ncol, hi, bnq[ncol]);
  acc[1] = gemm_tile(a, wb16 + (size_t)1 * 16384, ncol, hi, bnk[ncol]);
  acc[2] = gemm_tile(a, wb16 + (size_t)2 * 16384, ncol, hi, bnv[ncol]);

  float* outs[3] = {nq, nk, nv};
  const size_t base = (size_t)(row0 + hi * 8) * Dn + ncol;
  if (row0 + 16 <= nrows) {                 // uniform: full tile, no guards
#pragma unroll
    for (int m = 0; m < 3; ++m)
#pragma unroll
      for (int r = 0; r < 8; ++r) outs[m][base + (size_t)r * Dn] = acc[m][r];
  } else {
#pragma unroll
    for (int m = 0; m < 3; ++m)
#pragma unroll
      for (int r = 0; r < 8; ++r)
        if (row0 + hi * 8 + r < nrows) outs[m][base + (size_t)r * Dn] = acc[m][r];
  }
}

// ---------- k2: edge Q/K/V + gathered node proj + per-head scores ------------
__global__ __launch_bounds__(256) void k_edge_qkv_score(
    const float* __restrict__ edges, const int* __restrict__ eidx, int E,
    const __bf16* __restrict__ wb16,
    const float* __restrict__ beq, const float* __restrict__ bek,
    const float* __restrict__ bev,
    const float* __restrict__ nq, const float* __restrict__ nk,
    const float* __restrict__ nv,
    float* __restrict__ Vbuf, float* __restrict__ score) {
  __shared__ __attribute__((aligned(16))) __bf16 ldsA[16 * Dn];
  __shared__ int s_src[16], s_dst[16];
  const int tid = threadIdx.x;
  const int e0 = blockIdx.x * 16;
  stage_tile_bf16(edges, e0, E, tid, ldsA);
  if (tid < 16) {
    int ge = e0 + tid; if (ge >= E) ge = E - 1;
    s_src[tid] = eidx[ge];
  } else if (tid < 32) {
    int t = tid - 16;
    int ge = e0 + t; if (ge >= E) ge = E - 1;
    s_dst[t] = eidx[(size_t)E + ge];
  }
  __syncthreads();

  const int wave = tid >> 5, lane = tid & 31;     // wave == head (DH==16)
  const int col = lane & 15, hi = lane >> 4;
  v16bf a[4];
  build_a_frags(ldsA, lane, a);
  const int ncol = wave * 16 + col;               // head*16 + dh

  v8f q = gemm_tile(a, wb16 + (size_t)3 * 16384, ncol, hi, beq[ncol]);
  v8f k = gemm_tile(a, wb16 + (size_t)4 * 16384, ncol, hi, bek[ncol]);
  v8f v = gemm_tile(a, wb16 + (size_t)5 * 16384, ncol, hi, bev[ncol]);

  const float scale = 0.25f;  // 1/sqrt(DH), DH = 16
  const bool full = (e0 + 16 <= E);               // uniform across block
#pragma unroll
  for (int r = 0; r < 8; ++r) {
    int row = r + hi * 8;
    int s = s_src[row], d = s_dst[row];
    float qq = q[r] + nq[(size_t)s * Dn + ncol];
    float kk = k[r] + nk[(size_t)d * Dn + ncol];
    float vv = v[r] + nv[(size_t)d * Dn + ncol];
    int ge = e0 + row;
    if (full) {
      __builtin_nontemporal_store(vv, Vbuf + (size_t)ge * Dn + ncol);
    } else if (ge < E) {
      Vbuf[(size_t)ge * Dn + ncol] = vv;
    }
    // head-w score for edge (e0+row): reduce over 16 cols within half-wave
    float p = qq * kk;
#pragma unroll
    for (int off = 1; off < 16; off <<= 1) p += __shfl_xor(p, off, 32);
    if (col == 0 && (full || ge < E)) score[(size_t)ge * Hh + wave] = p * scale;
  }
}

// ---------- k3: segment max over edges grouped by src ------------------------
__global__ __launch_bounds__(256) void k_seg_max(
    const float* __restrict__ score, const int* __restrict__ src, int E,
    unsigned* __restrict__ mkey) {
  int i = blockIdx.x * 256 + threadIdx.x;
  if (i >= E * Hh) return;
  int e = i >> 3, h = i & 7;
  atomicMax(&mkey[(size_t)src[e] * Hh + h], fkey(score[i]));
}

// ---------- k4: exp + segment sum -------------------------------------------
__global__ __launch_bounds__(256) void k_seg_expsum(
    float* __restrict__ score, const int* __restrict__ src, int E,
    const unsigned* __restrict__ mkey, float* __restrict__ denom) {
  int i = blockIdx.x * 256 + threadIdx.x;
  if (i >= E * Hh) return;
  int e = i >> 3, h = i & 7;
  size_t gi = (size_t)src[e] * Hh + h;
  float ex = __expf(score[i] - funkey(mkey[gi]));
  score[i] = ex;                       // reuse score buffer to hold exp
  atomicAdd(&denom[gi], ex);
}

// ---------- k5: weighted scatter-add of V into agg (4-wide) ------------------
__global__ __launch_bounds__(256) void k_aggregate(
    const float* __restrict__ score, const float* __restrict__ denom,
    const int* __restrict__ src, int E,
    const float* __restrict__ Vbuf, float* __restrict__ agg) {
  long long i = (long long)blockIdx.x * 256 + threadIdx.x;
  if (i >= (long long)E * 32) return;          // 4 features per thread
  int e = (int)(i >> 5);
  int f4 = (int)(i & 31) << 2;                 // 0,4,...,124 (within one head)
  int h = f4 >> 4;
  int s = src[e];
  float w = score[(size_t)e * Hh + h] / denom[(size_t)s * Hh + h];
  v4f v = __builtin_nontemporal_load((const v4f*)(Vbuf + (size_t)e * Dn + f4));
  float* ag = agg + (size_t)s * Dn + f4;
  atomicAdd(ag + 0, w * v[0]);
  atomicAdd(ag + 1, w * v[1]);
  atomicAdd(ag + 2, w * v[2]);
  atomicAdd(ag + 3, w * v[3]);
}

// ---------- k6: output projection out = agg @ Wo.T + bo ----------------------
__global__ __launch_bounds__(256) void k_out_proj(
    const float* __restrict__ agg, const __bf16* __restrict__ wb16,
    const float* __restrict__ bo, float* __restrict__ out, int nrows) {
  __shared__ __attribute__((aligned(16))) __bf16 ldsA[16 * Dn];
  const int tid = threadIdx.x;
  const int row0 = blockIdx.x * 16;
  stage_tile_bf16(agg, row0, nrows, tid, ldsA);
  __syncthreads();
  const int wave = tid >> 5, lane = tid & 31;
  const int col = lane & 15, hi = lane >> 4;
  v16bf a[4];
  build_a_frags(ldsA, lane, a);
  const int ncol = wave * 16 + col;
  v8f c = gemm_tile(a, wb16 + (size_t)6 * 16384, ncol, hi, bo[ncol]);
  const size_t base = (size_t)(row0 + hi * 8) * Dn + ncol;
  if (row0 + 16 <= nrows) {
#pragma unroll
    for (int r = 0; r < 8; ++r) out[base + (size_t)r * Dn] = c[r];
  } else {
#pragma unroll
    for (int r = 0; r < 8; ++r)
      if (row0 + hi * 8 + r < nrows) out[base + (size_t)r * Dn] = c[r];
  }
}

// ---------- host ------------------------------------------------------------
extern "C" void kernel_launch(void* const* d_in, const int* in_sizes, int n_in,
                              void* d_out, int out_size, void* d_ws, size_t ws_size,
                              hipStream_t stream) {
  const float* nodes = (const float*)d_in[0];
  const float* edges = (const float*)d_in[1];
  const int*   eidx  = (const int*)d_in[2];
  const float* Wnq = (const float*)d_in[3];  const float* bnq = (const float*)d_in[4];
  const float* Wnk = (const float*)d_in[5];  const float* bnk = (const float*)d_in[6];
  const float* Wnv = (const float*)d_in[7];  const float* bnv = (const float*)d_in[8];
  const float* Weq = (const float*)d_in[9];  const float* beq = (const float*)d_in[10];
  const float* Wek = (const float*)d_in[11]; const float* bek = (const float*)d_in[12];
  const float* Wev = (const float*)d_in[13]; const float* bev = (const float*)d_in[14];
  const float* Wo  = (const float*)d_in[15]; const float* bo  = (const float*)d_in[16];

  const int N = in_sizes[0] / Dn;     // 50000
  const int E = in_sizes[2] / 2;      // 800000

  // workspace layout (256B-aligned slabs)
  char* ws = (char*)d_ws;
  size_t off = 0;
  auto carve = [&](size_t bytes) { void* p = ws + off; off = (off + bytes + 255) & ~(size_t)255; return p; };
  __bf16*  wb16  = (__bf16*)carve((size_t)7 * 16384 * sizeof(__bf16));
  float*   nq    = (float*)carve((size_t)N * Dn * 4);
  float*   nk    = (float*)carve((size_t)N * Dn * 4);
  float*   nv    = (float*)carve((size_t)N * Dn * 4);
  float*   Vbuf  = (float*)carve((size_t)E * Dn * 4);
  float*   score = (float*)carve((size_t)E * Hh * 4);
  unsigned* mkey = (unsigned*)carve((size_t)N * Hh * 4);
  float*   denom = (float*)carve((size_t)N * Hh * 4);
  float*   agg   = (float*)carve((size_t)N * Dn * 4);

  // zero the atomic targets (mkey key 0 < key(any float); denom/agg = 0)
  size_t zbytes = (size_t)((char*)agg - (char*)mkey) + (size_t)N * Dn * 4;
  hipMemsetAsync(mkey, 0, zbytes, stream);

  k_cvt_weights<<<(7 * 16384) / 256, 256, 0, stream>>>(Wnq, Wnk, Wnv, Weq, Wek, Wev, Wo, wb16);
  k_node_proj<<<(N + 15) / 16, 256, 0, stream>>>(nodes, wb16, bnq, bnk, bnv, nq, nk, nv, N);
  k_edge_qkv_score<<<(E + 15) / 16, 256, 0, stream>>>(edges, eidx, E, wb16, beq, bek, bev,
                                                      nq, nk, nv, Vbuf, score);
  k_seg_max<<<(E * Hh + 255) / 256, 256, 0, stream>>>(score, eidx, E, mkey);
  k_seg_expsum<<<(E * Hh + 255) / 256, 256, 0, stream>>>(score, eidx, E, mkey, denom);
  long long aggThreads = (long long)E * 32;
  k_aggregate<<<(unsigned)((aggThreads + 255) / 256), 256, 0, stream>>>(score, denom, eidx, E, Vbuf, agg);
  k_out_proj<<<(N + 15) / 16, 256, 0, stream>>>(agg, wb16, bo, (float*)d_out, N);
}
